// MaskMultiHeadAttention_62697932587329
// MI455X (gfx1250) — matrix operand order repaired and checked by reading, not compiled, not run on hardware
//
#include <hip/hip_runtime.h>

#define CC 64
#define TT 1024
#define EMB 1024
#define NH 16
#define PD 64
#define WPB 4          // waves per block
#define BLOCK_THREADS (WPB * 32)
#define NBLOCKS 4096
#define NTOK (CC * TT)

typedef __attribute__((ext_vector_type(16))) _Float16 v16h;
typedef __attribute__((ext_vector_type(8)))  _Float16 v8h;
typedef __attribute__((ext_vector_type(4)))  _Float16 v4h;
typedef __attribute__((ext_vector_type(8)))  float    v8f;

__device__ __forceinline__ void wave_lds_fence() {
    __builtin_amdgcn_wave_barrier();
    asm volatile("s_wait_dscnt 0" ::: "memory");
    __builtin_amdgcn_wave_barrier();
}

__global__ __launch_bounds__(BLOCK_THREADS)
void entmax_mha_kernel(const float* __restrict__ y,
                       const float* __restrict__ Wq, const float* __restrict__ bq,
                       const float* __restrict__ Wk, const float* __restrict__ bk,
                       const float* __restrict__ Wv, const float* __restrict__ bv,
                       float* __restrict__ out)
{
    // Weights as f16 (B-matrix source), biases f32
    __shared__ __align__(32) _Float16 sW[3][PD][PD];     // 24 KB, [proj][dout][din]
    __shared__ float                  sB[3][PD];          // 768 B
    // Per-wave scratch
    __shared__ __align__(32) _Float16 sX [WPB][EMB];      // masked x, f16
    __shared__ __align__(32) _Float16 sQ [WPB][NH][PD];
    __shared__ __align__(32) _Float16 sK [WPB][NH][PD];
    __shared__ __align__(32) _Float16 sVT[WPB][PD][NH];   // v transposed [d][g]
    __shared__ float                  sS [WPB][NH][NH];   // scores f32
    __shared__ __align__(32) _Float16 sA [WPB][NH][NH];   // attn f16

    const int tid  = threadIdx.x;
    const int lane = tid & 31;
    const int wv   = tid >> 5;
    const int lo   = lane & 15;
    const int hi   = lane >> 4;   // 0: lanes 0-15, 1: lanes 16-31

    // ---- stage weights (block-wide, once) ----
    {
        const float* Wsrc[3] = {Wq, Wk, Wv};
        const float* bsrc[3] = {bq, bk, bv};
        #pragma unroll
        for (int p = 0; p < 3; ++p) {
            _Float16* dst = &sW[p][0][0];
            const float* src = Wsrc[p];
            for (int i = tid; i < PD * PD; i += BLOCK_THREADS)
                dst[i] = (_Float16)src[i];
            if (tid < PD) sB[p][tid] = bsrc[p][tid];
        }
    }
    __syncthreads();

    const int nWaves = gridDim.x * WPB;
    for (int tok = blockIdx.x * WPB + wv; tok < NTOK; tok += nWaves) {
        const int t = tok & (TT - 1);
        const float* row = y + (size_t)tok * EMB;

        // prefetch next token's row (one cacheline per lane covers 4 KB)
        if (tok + nWaves < NTOK)
            __builtin_prefetch(y + (size_t)(tok + nWaves) * EMB + lane * 32, 0, 1);

        // ---- stage masked x row as f16 (mask: keep e < t) ----
        #pragma unroll
        for (int j = 0; j < 8; ++j) {
            int e = j * 128 + lane * 4;
            float4 f = *(const float4*)(row + e);
            v4h h;
            h[0] = (e + 0 < t) ? (_Float16)f.x : (_Float16)0.f;
            h[1] = (e + 1 < t) ? (_Float16)f.y : (_Float16)0.f;
            h[2] = (e + 2 < t) ? (_Float16)f.z : (_Float16)0.f;
            h[3] = (e + 3 < t) ? (_Float16)f.w : (_Float16)0.f;
            *(v4h*)&sX[wv][e] = h;
        }
        wave_lds_fence();

        // ---- A-fragments of x (16x32 f16 per k-step; lane<16: K=0..15) ----
        v16h ax[2];
        #pragma unroll
        for (int ks = 0; ks < 2; ++ks) {
            int base = lo * PD + ks * 32 + hi * 16;
            v8h c0 = *(const v8h*)&sX[wv][base];
            v8h c1 = *(const v8h*)&sX[wv][base + 8];
            #pragma unroll
            for (int i = 0; i < 8; ++i) { ax[ks][i] = c0[i]; ax[ks][i + 8] = c1[i]; }
        }

        // ---- q, k, v projections: D[h][dout] = sum_d x[h][d] * W[dout][d] + b[dout] ----
        #pragma unroll
        for (int p = 0; p < 3; ++p) {
            #pragma unroll
            for (int nt = 0; nt < 4; ++nt) {
                const int dout = nt * 16 + lo;
                const float bb = sB[p][dout];
                v8f acc = {bb, bb, bb, bb, bb, bb, bb, bb};
                #pragma unroll
                for (int ks = 0; ks < 2; ++ks) {
                    // B 32x16: lane<16 holds K=ks*32+0..15 of col N=lo (row dout of W)
                    const v16h bf = *(const v16h*)&sW[p][dout][ks * 32 + hi * 16];
                    acc = __builtin_amdgcn_wmma_f32_16x16x32_f16(
                              false, ax[ks], false, bf, (short)0, acc, false, false);
                }
                if (p == 0) {
                    #pragma unroll
                    for (int r = 0; r < 8; ++r) sQ[wv][r + 8 * hi][dout] = (_Float16)acc[r];
                } else if (p == 1) {
                    #pragma unroll
                    for (int r = 0; r < 8; ++r) sK[wv][r + 8 * hi][dout] = (_Float16)acc[r];
                } else {
                    v8h pk;
                    #pragma unroll
                    for (int r = 0; r < 8; ++r) pk[r] = (_Float16)acc[r];
                    *(v8h*)&sVT[wv][dout][8 * hi] = pk;   // vT[d][g-chunk], contiguous
                }
            }
        }
        wave_lds_fence();

        // ---- scores[h][g] = sum_d q[h][d] * k[g][d] ----
        {
            v8f acc = {0.f, 0.f, 0.f, 0.f, 0.f, 0.f, 0.f, 0.f};
            #pragma unroll
            for (int ks = 0; ks < 2; ++ks) {
                const _Float16* q = &sQ[wv][0][0];
                int base = lo * PD + ks * 32 + hi * 16;
                v8h c0 = *(const v8h*)(q + base);
                v8h c1 = *(const v8h*)(q + base + 8);
                v16h qa;
                #pragma unroll
                for (int i = 0; i < 8; ++i) { qa[i] = c0[i]; qa[i + 8] = c1[i]; }
                const v16h kb = *(const v16h*)&sK[wv][lo][ks * 32 + hi * 16];
                acc = __builtin_amdgcn_wmma_f32_16x16x32_f16(
                          false, qa, false, kb, (short)0, acc, false, false);
            }
            #pragma unroll
            for (int r = 0; r < 8; ++r) sS[wv][r + 8 * hi][lo] = acc[r];
        }
        wave_lds_fence();

        // ---- exact 1.5-entmax per row, one row per lane (lanes 0..15) ----
        if (lane < 16) {
            float z[16];
            #pragma unroll
            for (int i = 0; i < 16; ++i)
                z[i] = sS[wv][lane][i] * 0.015625f;   // (1/sqrt(1024)) * (1/2)
            float zmax = z[0];
            #pragma unroll
            for (int i = 1; i < 16; ++i) zmax = fmaxf(zmax, z[i]);
            #pragma unroll
            for (int i = 0; i < 16; ++i) z[i] -= zmax;

            float srt[16];
            #pragma unroll
            for (int i = 0; i < 16; ++i) srt[i] = z[i];
            // bitonic sort ascending, fully unrolled (stays in registers)
            #pragma unroll
            for (int kk = 2; kk <= 16; kk <<= 1) {
                #pragma unroll
                for (int jj = kk >> 1; jj > 0; jj >>= 1) {
                    #pragma unroll
                    for (int i = 0; i < 16; ++i) {
                        int l = i ^ jj;
                        if (l > i) {
                            float a = srt[i], b = srt[l];
                            bool up = ((i & kk) == 0);
                            bool sw = up ? (a > b) : (a < b);
                            srt[i] = sw ? b : a;
                            srt[l] = sw ? a : b;
                        }
                    }
                }
            }
            float cum = 0.f, cms = 0.f, tau[16];
            int cnt = 0;
            #pragma unroll
            for (int k = 0; k < 16; ++k) {
                float v = srt[15 - k];            // descending order
                cum += v; cms += v * v;
                float kk1  = (float)(k + 1);
                float mean = cum / kk1;
                float ss   = cms - cum * mean;
                float del  = (1.0f - ss) / kk1;
                del = del > 0.f ? del : 0.f;
                float tk = mean - __builtin_sqrtf(del);
                tau[k] = tk;
                cnt += (tk <= v) ? 1 : 0;
            }
            float taus = tau[0];
            #pragma unroll
            for (int k = 1; k < 16; ++k) if (cnt - 1 == k) taus = tau[k];

            v8h a0, a1;
            #pragma unroll
            for (int i = 0; i < 16; ++i) {
                float d = z[i] - taus;
                d = d > 0.f ? d : 0.f;
                float w = d * d;
                if (i < 8) a0[i] = (_Float16)w; else a1[i - 8] = (_Float16)w;
            }
            *(v8h*)&sA[wv][lane][0] = a0;
            *(v8h*)&sA[wv][lane][8] = a1;
        }
        wave_lds_fence();

        // ---- out[h][d] = sum_g attn[h][g] * v[g][d]  (K=16 zero-padded to 32) ----
        v16h af;
        {
            v16h c = *(const v16h*)&sA[wv][lo][0];
            #pragma unroll
            for (int i = 0; i < 16; ++i) af[i] = hi ? (_Float16)0.f : c[i];
        }
        float* orow = out + (size_t)tok * EMB;
        #pragma unroll
        for (int nt = 0; nt < 4; ++nt) {
            const int dcol = nt * 16 + lo;
            const v16h vb = *(const v16h*)&sVT[wv][dcol][0];  // K rows 16..31 don't matter: A pad=0
            v8f acc = {0.f, 0.f, 0.f, 0.f, 0.f, 0.f, 0.f, 0.f};
            acc = __builtin_amdgcn_wmma_f32_16x16x32_f16(
                      false, af, false, vb, (short)0, acc, false, false);
            #pragma unroll
            for (int r = 0; r < 8; ++r)
                orow[(r + 8 * hi) * PD + dcol] = acc[r];
        }
    }
}

extern "C" void kernel_launch(void* const* d_in, const int* in_sizes, int n_in,
                              void* d_out, int out_size, void* d_ws, size_t ws_size,
                              hipStream_t stream) {
    (void)in_sizes; (void)n_in; (void)out_size; (void)d_ws; (void)ws_size;
    const float* y  = (const float*)d_in[0];
    const float* Wq = (const float*)d_in[1];
    const float* bq = (const float*)d_in[2];
    const float* Wk = (const float*)d_in[3];
    const float* bk = (const float*)d_in[4];
    const float* Wv = (const float*)d_in[5];
    const float* bv = (const float*)d_in[6];
    float* out = (float*)d_out;
    entmax_mha_kernel<<<NBLOCKS, BLOCK_THREADS, 0, stream>>>(y, Wq, bq, Wk, bk, Wv, bv, out);
}
